// RVQVAE_65532611003015
// MI455X (gfx1250) — compile-verified, matrix operand order checked by reading, not run on hardware
//
#include <hip/hip_runtime.h>
#include <hip/hip_bf16.h>
#include <math.h>

// ---------------------------------------------------------------------------
// RVQ-VAE forward for gfx1250 (MI455X). All GEMM-shaped work (attention
// projections / scores / context, conv-as-implicit-GEMM, codebook distances)
// runs through v_wmma_f32_16x16x32_f16. Wave32.
//   - gemm kernel: 32x32 tile per wave, guard-free (all call shapes are
//     multiples of 32), A / B(T) fragments loaded as float4 (global_load_b128).
//   - conv kernel: 16x32 tile per wave, branch-free clamped-address gathers.
// ---------------------------------------------------------------------------

typedef _Float16 f16_t;
typedef __attribute__((ext_vector_type(16))) _Float16 v16h;
typedef __attribute__((ext_vector_type(8)))  float    v8f;

#define GN_GROUPS 32

static __device__ __forceinline__ v16h pack16(float4 x0, float4 x1, float4 x2, float4 x3)
{
    v16h v;
    v[0]  = (f16_t)x0.x; v[1]  = (f16_t)x0.y; v[2]  = (f16_t)x0.z; v[3]  = (f16_t)x0.w;
    v[4]  = (f16_t)x1.x; v[5]  = (f16_t)x1.y; v[6]  = (f16_t)x1.z; v[7]  = (f16_t)x1.w;
    v[8]  = (f16_t)x2.x; v[9]  = (f16_t)x2.y; v[10] = (f16_t)x2.z; v[11] = (f16_t)x2.w;
    v[12] = (f16_t)x3.x; v[13] = (f16_t)x3.y; v[14] = (f16_t)x3.z; v[15] = (f16_t)x3.w;
    return v;
}

// Load a 16-half fragment for logical row `row` (A layout; also B^T layout):
// elements k0+khalf+[0..7] and k0+khalf+16+[0..7], contiguous, 32B aligned.
static __device__ __forceinline__ v16h load_frag_rowmajor(const float* __restrict__ P,
                                                          long rowbase, int k0, int khalf)
{
    const float* p = P + rowbase + k0 + khalf;
    const float4* q = reinterpret_cast<const float4*>(p);
    float4 x0 = q[0];
    float4 x1 = q[1];
    const float4* q2 = reinterpret_cast<const float4*>(p + 16);
    float4 x2 = q2[0];
    float4 x3 = q2[1];
    return pack16(x0, x1, x2, x3);
}

// ---------------------------------------------------------------------------
// Generic GEMM:  C[M,N] = alpha * A[M,K] @ B  (+bias[n]) (+C if accum)
//   bT=1: B stored as [N,K]  -> A @ B^T   (vectorized fragment loads)
//   bT=0: B stored as [K,N]  -> A @ B     (strided scalar loads, branch-free)
// REQUIRES: M%32==0, N%32==0, K%32==0, lda/ldb multiples of 8. (All callers
// in this network satisfy this.)  One 32x32 tile per wave; block = 2 waves.
// ---------------------------------------------------------------------------
__global__ __launch_bounds__(64) void gemm_wmma_kernel(
    const float* __restrict__ A, int lda,
    const float* __restrict__ Bm, int ldb,
    float* __restrict__ C, int ldc,
    int M, int N, int K, int bT, float alpha,
    const float* __restrict__ bias, int accum)
{
    const int wave = threadIdx.x >> 5;
    const int lane = threadIdx.x & 31;
    const int m0 = blockIdx.y << 5;
    const int n0 = (blockIdx.x * 2 + wave) << 5;
    if (n0 >= N) return;                 // wave-uniform

    const int r     = lane & 15;
    const int khalf = (lane >> 4) << 3;  // lanes 16-31 hold K+8 / K+24

    const long arow0 = (long)(m0 + r) * lda;
    const long arow1 = (long)(m0 + 16 + r) * lda;
    const long brow0 = (long)(n0 + r) * ldb;         // bT=1 path
    const long brow1 = (long)(n0 + 16 + r) * ldb;

    v8f acc00 = {}, acc01 = {}, acc10 = {}, acc11 = {};

    for (int k0 = 0; k0 < K; k0 += 32) {
        v16h a0 = load_frag_rowmajor(A, arow0, k0, khalf);
        v16h a1 = load_frag_rowmajor(A, arow1, k0, khalf);
        v16h b0, b1;
        if (bT) {
            b0 = load_frag_rowmajor(Bm, brow0, k0, khalf);
            b1 = load_frag_rowmajor(Bm, brow1, k0, khalf);
        } else {
#pragma unroll
            for (int j = 0; j < 16; ++j) {
                int k = k0 + khalf + j + ((j >> 3) << 3);
                b0[j] = (f16_t)Bm[(long)k * ldb + (n0 + r)];
                b1[j] = (f16_t)Bm[(long)k * ldb + (n0 + 16 + r)];
            }
        }
        acc00 = __builtin_amdgcn_wmma_f32_16x16x32_f16(false, a0, false, b0, (short)0, acc00, false, false);
        acc01 = __builtin_amdgcn_wmma_f32_16x16x32_f16(false, a0, false, b1, (short)0, acc01, false, false);
        acc10 = __builtin_amdgcn_wmma_f32_16x16x32_f16(false, a1, false, b0, (short)0, acc10, false, false);
        acc11 = __builtin_amdgcn_wmma_f32_16x16x32_f16(false, a1, false, b1, (short)0, acc11, false, false);
    }

    const int cbase = lane & 15;
    const int rbase = (lane >> 4) << 3;
    v8f* accs[4] = { &acc00, &acc01, &acc10, &acc11 };
#pragma unroll
    for (int t = 0; t < 4; ++t) {
        const int mt = (t >> 1) << 4;
        const int nt = (t & 1) << 4;
        const int col = n0 + nt + cbase;
        const float badd = bias ? bias[col] : 0.0f;
#pragma unroll
        for (int i = 0; i < 8; ++i) {
            int row = m0 + mt + rbase + i;
            long off = (long)row * ldc + col;
            float v = (*accs[t])[i] * alpha + badd;
            if (accum) v += C[off];
            C[off] = v;
        }
    }
}

// ---------------------------------------------------------------------------
// Implicit-GEMM convolution (NCHW / OIHW), optional lhs-dilation (dil) for
// transposed convs.  M = B*Ho*Wo (multiple of 16), N = Co, K = Ci*kh*kw.
// 16x32 tile per wave; branch-free clamped gathers (no exec-mask branches).
// ---------------------------------------------------------------------------
__global__ __launch_bounds__(64) void conv_wmma_kernel(
    const float* __restrict__ X, const float* __restrict__ Wt,
    const float* __restrict__ bias, float* __restrict__ Y,
    int Bn, int Ci, int Hin, int Win, int Co, int Ho, int Wo,
    int kh, int kw, int stride, int pad_h, int pad_w, int dil, int accum)
{
    const int wave = threadIdx.x >> 5;
    const int lane = threadIdx.x & 31;
    const int M = Bn * Ho * Wo;
    const int K = Ci * kh * kw;
    const int m0 = blockIdx.y << 4;
    const int n0 = (blockIdx.x * 2 + wave) << 5;
    if (m0 >= M || n0 >= Co) return;

    const int r     = lane & 15;
    const int khalf = (lane >> 4) << 3;
    const int khkw  = kh * kw;

    // decode this lane's output coordinate (always valid: M % 16 == 0)
    const int mrow = m0 + r;
    const int ow = mrow % Wo;
    const int tt = mrow / Wo;
    const int oh = tt % Ho;
    const int bb = tt / Ho;

    const int nc0 = n0 + r;
    const int nc1 = n0 + 16 + r;
    const long wrow0 = (long)(nc0 < Co ? nc0 : Co - 1) * K;
    const long wrow1 = (long)(nc1 < Co ? nc1 : Co - 1) * K;

    v8f acc0 = {}, acc1 = {};

    for (int k0 = 0; k0 < K; k0 += 32) {
        v16h a, b0, b1;
#pragma unroll
        for (int j = 0; j < 16; ++j) {
            const int k   = k0 + khalf + j + ((j >> 3) << 3);
            const bool kok = (k < K);
            const int kc  = kok ? k : 0;

            // ---- A: gathered input element, clamped address + select ----
            const int ci  = kc / khkw;
            const int rem = kc - ci * khkw;
            const int rr  = rem / kw;
            const int ss  = rem - rr * kw;
            const int th = oh * stride - pad_h + rr;
            const int tw = ow * stride - pad_w + ss;
            int ih = th, iw = tw;
            bool ok = kok;
            if (dil != 1) {
                ok = ok && (th >= 0) && (tw >= 0) && (th % dil == 0) && (tw % dil == 0);
                ih = th / dil; iw = tw / dil;
            }
            ok = ok && (ih >= 0) && (ih < Hin) && (iw >= 0) && (iw < Win);
            long off = ok ? ((((long)bb * Ci + ci) * Hin + ih) * Win + iw) : 0;
            float av = X[off];
            a[j] = (f16_t)(ok ? av : 0.0f);

            // ---- B: weights, contiguous in k, clamped + select ----
            float bv0 = Wt[wrow0 + kc];
            float bv1 = Wt[wrow1 + kc];
            b0[j] = (f16_t)((kok && nc0 < Co) ? bv0 : 0.0f);
            b1[j] = (f16_t)((kok && nc1 < Co) ? bv1 : 0.0f);
        }
        acc0 = __builtin_amdgcn_wmma_f32_16x16x32_f16(false, a, false, b0, (short)0, acc0, false, false);
        acc1 = __builtin_amdgcn_wmma_f32_16x16x32_f16(false, a, false, b1, (short)0, acc1, false, false);
    }

    const int cbase = lane & 15;
    const int rbase = (lane >> 4) << 3;
#pragma unroll
    for (int t = 0; t < 2; ++t) {
        const int col = n0 + (t << 4) + cbase;
        if (col < Co) {
            const float badd = bias[col];
            const v8f& acc = t ? acc1 : acc0;
#pragma unroll
            for (int i = 0; i < 8; ++i) {
                int row = m0 + rbase + i;
                int ow2 = row % Wo; int t2 = row / Wo; int oh2 = t2 % Ho; int b2 = t2 / Ho;
                long off = (((long)b2 * Co + col) * Ho + oh2) * Wo + ow2;
                float v = acc[i] + badd;
                if (accum) v += Y[off];
                Y[off] = v;
            }
        }
    }
}

// ---------------------------------------------------------------------------
// GroupNorm (32 groups) + optional SiLU, optional transposed (B,L,C) output.
// ---------------------------------------------------------------------------
__global__ void groupnorm_kernel(const float* __restrict__ X, float* __restrict__ Y,
                                 const float* __restrict__ gam, const float* __restrict__ bet,
                                 int C, int L, int do_silu, int tpose)
{
    const int grp = blockIdx.x;
    const int bb  = blockIdx.y;
    const int cpg = C / GN_GROUPS;
    const long base = ((long)bb * C + (long)grp * cpg) * L;
    const long n    = (long)cpg * L;

    float s = 0.f, s2 = 0.f;
    for (long i = threadIdx.x; i < n; i += blockDim.x) {
        float v = X[base + i]; s += v; s2 += v * v;
    }
    __shared__ float sh1[256], sh2[256];
    sh1[threadIdx.x] = s; sh2[threadIdx.x] = s2; __syncthreads();
    for (int st = 128; st > 0; st >>= 1) {
        if ((int)threadIdx.x < st) {
            sh1[threadIdx.x] += sh1[threadIdx.x + st];
            sh2[threadIdx.x] += sh2[threadIdx.x + st];
        }
        __syncthreads();
    }
    const float mean = sh1[0] / (float)n;
    const float var  = sh2[0] / (float)n - mean * mean;
    const float inv  = rsqrtf(var + 1e-5f);

    for (long i = threadIdx.x; i < n; i += blockDim.x) {
        int  cl = (int)(i / L);
        long l  = i - (long)cl * L;
        int  c  = grp * cpg + cl;
        float v = (X[base + i] - mean) * inv * gam[c] + bet[c];
        if (do_silu) v = v / (1.f + __expf(-v));
        if (tpose) Y[((long)bb * L + l) * C + c] = v;
        else       Y[base + i] = v;
    }
}

// Row softmax (in place). grid = rows, block = 256.
__global__ void softmax_rows_kernel(float* __restrict__ S, int ncol)
{
    float* p = S + (long)blockIdx.x * ncol;
    __shared__ float red[256];
    float m = -3.4e38f;
    for (int i = threadIdx.x; i < ncol; i += blockDim.x) m = fmaxf(m, p[i]);
    red[threadIdx.x] = m; __syncthreads();
    for (int st = 128; st > 0; st >>= 1) {
        if ((int)threadIdx.x < st) red[threadIdx.x] = fmaxf(red[threadIdx.x], red[threadIdx.x + st]);
        __syncthreads();
    }
    m = red[0]; __syncthreads();
    float s = 0.f;
    for (int i = threadIdx.x; i < ncol; i += blockDim.x) {
        float e = __expf(p[i] - m); p[i] = e; s += e;
    }
    red[threadIdx.x] = s; __syncthreads();
    for (int st = 128; st > 0; st >>= 1) {
        if ((int)threadIdx.x < st) red[threadIdx.x] += red[threadIdx.x + st];
        __syncthreads();
    }
    const float inv = 1.f / red[0];
    for (int i = threadIdx.x; i < ncol; i += blockDim.x) p[i] *= inv;
}

// (B,C,L) -> (B,L,C)
__global__ void nchw2nhwc_kernel(const float* __restrict__ X, float* __restrict__ Y,
                                 int C, int L, long total)
{
    long i = (long)blockIdx.x * blockDim.x + threadIdx.x;
    if (i >= total) return;
    long l = i % L; long t = i / L; long c = t % C; long b = t / C;
    Y[(b * L + l) * C + c] = X[i];
}

// (B,L,C) -> (B,C,L)
__global__ void nhwc2nchw_kernel(const float* __restrict__ X, float* __restrict__ Y,
                                 int C, int L, long total)
{
    long i = (long)blockIdx.x * blockDim.x + threadIdx.x;
    if (i >= total) return;
    long l = i % L; long t = i / L; long c = t % C; long b = t / C;
    Y[i] = X[(b * L + l) * C + c];
}

// X[b,c,l] += T[b,l,c]   (attention residual)
__global__ void add_transposed_kernel(float* __restrict__ X, const float* __restrict__ T,
                                      int C, int L, long total)
{
    long i = (long)blockIdx.x * blockDim.x + threadIdx.x;
    if (i >= total) return;
    long l = i % L; long t = i / L; long c = t % C; long b = t / C;
    X[i] += T[(b * L + l) * C + c];
}

// Crop rows: (B,C,Hsrc,W) -> (B,C,Hdst,W), Hdst <= Hsrc
__global__ void crop_rows_kernel(const float* __restrict__ S, float* __restrict__ D,
                                 int C, int Hsrc, int Hdst, int W, long total)
{
    long i = (long)blockIdx.x * blockDim.x + threadIdx.x;
    if (i >= total) return;
    long w = i % W; long t = i / W; long h = t % Hdst; t /= Hdst;
    long c = t % C; long b = t / C;
    D[i] = S[(((b * C + c) * Hsrc) + h) * W + w];
}

__global__ void zero_kernel(float* __restrict__ p, long n)
{
    long i = (long)blockIdx.x * blockDim.x + threadIdx.x;
    if (i < n) p[i] = 0.f;
}

// ||cb[n]||^2 for each codebook row
__global__ void cbnorm_kernel(const float* __restrict__ cb, float* __restrict__ out,
                              int Kcb, int Cz)
{
    int n = blockIdx.x * blockDim.x + threadIdx.x;
    if (n >= Kcb) return;
    float s = 0.f;
    for (int c = 0; c < Cz; ++c) { float v = cb[(long)n * Cz + c]; s += v * v; }
    out[n] = s;
}

// Per row m: idx = argmin_n (||cb_n||^2 - 2*dot[m,n]); qacc += cb[idx];
// xf -= cb[idx]; losses += mean((xf_old - cb[idx])^2). block = 128, grid = M.
__global__ void argmin_update_kernel(const float* __restrict__ dotm,
                                     const float* __restrict__ cbn,
                                     const float* __restrict__ cb,
                                     float* __restrict__ xf, float* __restrict__ qacc,
                                     float* __restrict__ loss_cb, float* __restrict__ loss_cm,
                                     int Kcb, int Cz, float inv_mc)
{
    const int m = blockIdx.x;
    __shared__ float bv[128]; __shared__ int bi[128];
    __shared__ float ps[128]; __shared__ int sidx;
    float best = 3.4e38f; int besti = 0;
    for (int n = threadIdx.x; n < Kcb; n += blockDim.x) {
        float d = cbn[n] - 2.f * dotm[(long)m * Kcb + n];
        if (d < best) { best = d; besti = n; }
    }
    bv[threadIdx.x] = best; bi[threadIdx.x] = besti; __syncthreads();
    for (int st = 64; st > 0; st >>= 1) {
        if ((int)threadIdx.x < st) {
            float ov = bv[threadIdx.x + st]; int oi = bi[threadIdx.x + st];
            if (ov < bv[threadIdx.x] || (ov == bv[threadIdx.x] && oi < bi[threadIdx.x])) {
                bv[threadIdx.x] = ov; bi[threadIdx.x] = oi;
            }
        }
        __syncthreads();
    }
    if (threadIdx.x == 0) sidx = bi[0];
    __syncthreads();
    const int idx = sidx;

    float part = 0.f;
    if ((int)threadIdx.x < Cz) {
        int c = threadIdx.x;
        float q = cb[(long)idx * Cz + c];
        long off = (long)m * Cz + c;
        float x = xf[off];
        float d = x - q;
        qacc[off] += q;
        xf[off] = d;
        part = d * d;
    }
    ps[threadIdx.x] = part; __syncthreads();
    for (int st = 64; st > 0; st >>= 1) {
        if ((int)threadIdx.x < st) ps[threadIdx.x] += ps[threadIdx.x + st];
        __syncthreads();
    }
    if (threadIdx.x == 0) {
        atomicAdd(loss_cb, ps[0] * inv_mc);
        atomicAdd(loss_cm, ps[0] * inv_mc);
    }
}

// ---------------------------------------------------------------------------
// Host orchestration
// ---------------------------------------------------------------------------
extern "C" void kernel_launch(void* const* d_in, const int* in_sizes, int n_in,
                              void* d_out, int out_size, void* d_ws, size_t ws_size,
                              hipStream_t stream)
{
    (void)in_sizes; (void)n_in; (void)out_size; (void)ws_size;
    auto IN = [&](int i) { return (const float*)d_in[i]; };

    // ---- workspace layout (floats) ----
    float* ws = (float*)d_ws;
    const long CAPe = 8519680;                 // (2,128,130,256) = largest tensor
    float* T0 = ws;
    float* T1 = T0 + CAPe;
    float* T2 = T1 + CAPe;
    float* T3 = T2 + CAPe;
    const long LC = 2L * 2048 * 256;           // (B,L,C) at mid resolution
    float* At   = T3 + CAPe;
    float* Aq   = At + LC;
    float* Ak   = Aq + LC;
    float* Av   = Ak + LC;
    float* Ao   = Av + LC;
    float* Atmp = Ao + LC;
    float* SC   = Atmp + LC;                   // 2048*2048 scores (per b,h)
    float* QXF  = SC + 4194304;                // 4096*64 residual (NHWC)
    float* QACC = QXF + 262144;                // 4096*64 quant accumulator
    float* QDOT = QACC + 262144;               // 4096*1024 dot products
    float* QCBN = QDOT + 4194304;              // 1024 codebook norms

    float* OUT = (float*)d_out;
    const long OUTN = 2L * 2 * 129 * 256;
    float* LCB = OUT + OUTN;                   // cb_loss scalar
    float* LCM = OUT + OUTN + 1;               // commit loss scalar

    // ---- launch helpers ----
    // gemm: requires M%32==0, N%32==0, K%32==0 (true for every call below)
    auto gemm = [&](const float* A, int lda, const float* Bm, int ldb,
                    float* Cc, int ldc, int M, int N, int K,
                    int bT, float alpha, const float* bias, int accum) {
        dim3 g((unsigned)((N + 63) / 64), (unsigned)(M / 32));
        gemm_wmma_kernel<<<g, 64, 0, stream>>>(A, lda, Bm, ldb, Cc, ldc,
                                               M, N, K, bT, alpha, bias, accum);
    };
    auto conv = [&](const float* X, int wi, float* Y, int Ci, int Hin, int Win,
                    int Co, int Ho, int Wo, int kh, int kw, int stride,
                    int ph, int pw, int dil, int accum) {
        int M = 2 * Ho * Wo;
        dim3 g((unsigned)((Co + 63) / 64), (unsigned)((M + 15) / 16));
        conv_wmma_kernel<<<g, 64, 0, stream>>>(X, IN(wi), IN(wi + 1), Y,
                                               2, Ci, Hin, Win, Co, Ho, Wo,
                                               kh, kw, stride, ph, pw, dil, accum);
    };
    auto gn = [&](const float* X, float* Y, int gi, int C, int L, int silu, int tp) {
        dim3 g(GN_GROUPS, 2);
        groupnorm_kernel<<<g, 256, 0, stream>>>(X, Y, IN(gi), IN(gi + 1), C, L, silu, tp);
    };
    // ResNet block: OUTb = conv_res(X) + c2(silu(gn2(c1(silu(gn1(X))))))
    auto resnet = [&](int pb, int Cin, int Cout, int H, int W,
                      const float* X, float* OUTb) {
        int L = H * W;
        gn(X, T1, pb + 0, Cin, L, 1, 0);
        conv(T1, pb + 2, T2, Cin, H, W, Cout, H, W, 3, 3, 1, 1, 1, 1, 0);
        gn(T2, T1, pb + 4, Cout, L, 1, 0);
        conv(X, pb + 8, OUTb, Cin, H, W, Cout, H, W, 1, 1, 1, 0, 0, 1, 0);
        conv(T1, pb + 6, OUTb, Cout, H, W, Cout, H, W, 3, 3, 1, 1, 1, 1, 1);
    };
    // Attention block, in-place on X (B,C,H,W)
    auto attn = [&](int pb, float* X, int C, int H, int W) {
        const int L = H * W;               // 2048
        const int BL = 2 * L;
        const int hd = C / 4;              // 64
        gn(X, At, pb + 0, C, L, 0, 1);     // -> (B,L,C)
        gemm(At, C, IN(pb + 2), C, Aq, C, BL, C, C, 1, 1.f, nullptr, 0);
        gemm(At, C, IN(pb + 3), C, Ak, C, BL, C, C, 1, 1.f, nullptr, 0);
        gemm(At, C, IN(pb + 4), C, Av, C, BL, C, C, 1, 1.f, nullptr, 0);
        const float sc = 1.0f / sqrtf((float)hd);
        for (int b = 0; b < 2; ++b)
            for (int h = 0; h < 4; ++h) {
                const float* Qp = Aq + (long)b * L * C + h * hd;
                const float* Kp = Ak + (long)b * L * C + h * hd;
                const float* Vp = Av + (long)b * L * C + h * hd;
                float*       Op = Ao + (long)b * L * C + h * hd;
                gemm(Qp, C, Kp, C, SC, L, L, L, hd, 1, sc, nullptr, 0);   // Q K^T
                softmax_rows_kernel<<<L, 256, 0, stream>>>(SC, L);
                gemm(SC, L, Vp, C, Op, C, L, hd, L, 0, 1.f, nullptr, 0);  // A V
            }
        gemm(Ao, C, IN(pb + 5), C, Atmp, C, BL, C, C, 1, 1.f, nullptr, 0); // O wo^T
        long tot = (long)2 * C * L;
        add_transposed_kernel<<<(unsigned)((tot + 255) / 256), 256, 0, stream>>>(X, Atmp, C, L, tot);
    };
    // Mid block: T0 -> T0 (uses T1..T3 as scratch)
    auto mid = [&](int pb, int C, int H, int W) {
        resnet(pb + 0, C, C, H, W, T0, T3);
        attn(pb + 10, T3, C, H, W);
        resnet(pb + 16, C, C, H, W, T3, T0);
    };

    // ================= pipeline =================
    // Input leaf order (depth-first, dict insertion order):
    //  0:x 1-2:conv_in 3-12:down1.res 13-14:down1.down 15-24:down2.res
    //  25-26:down2.down 27-52:mid_enc 53-54:enc_norm 55-56:enc_out
    //  57-58:pre_quant 59:codebooks 60-61:post_quant 62-63:dec_in
    //  64-89:mid_dec 90-91:up1.up 92-101:up1.res 102-103:up2.up
    //  104-113:up2.res 114-115:dec_norm 116-117:conv_out

    conv(IN(0), 1, T0, 2, 129, 256, 64, 129, 256, 3, 3, 1, 1, 1, 1, 0);   // conv_in

    resnet(3, 64, 128, 129, 256, T0, T3);                                 // down1.res
    conv(T3, 13, T0, 128, 129, 256, 128, 64, 128, 4, 4, 2, 1, 1, 1, 0);   // down1.down

    resnet(15, 128, 256, 64, 128, T0, T3);                                // down2.res
    conv(T3, 25, T0, 256, 64, 128, 256, 32, 64, 4, 4, 2, 1, 1, 1, 0);     // down2.down

    mid(27, 256, 32, 64);                                                 // mid_enc

    gn(T0, T1, 53, 256, 2048, 1, 0);                                      // enc_norm+silu
    conv(T1, 55, T2, 256, 32, 64, 64, 32, 64, 3, 3, 1, 1, 1, 1, 0);       // enc_out
    conv(T2, 57, T3, 64, 32, 64, 64, 32, 64, 1, 1, 1, 0, 0, 1, 0);        // pre_quant -> z

    // ---- residual VQ (4 stages) ----
    nchw2nhwc_kernel<<<(262144 + 255) / 256, 256, 0, stream>>>(T3, QXF, 64, 2048, 262144L);
    zero_kernel<<<(262144 + 255) / 256, 256, 0, stream>>>(QACC, 262144L);
    zero_kernel<<<1, 2, 0, stream>>>(LCB, 2L);
    const float inv_mc = 1.0f / (4096.0f * 64.0f);
    for (int qi = 0; qi < 4; ++qi) {
        const float* cb = IN(59) + (long)qi * 1024 * 64;
        cbnorm_kernel<<<(1024 + 127) / 128, 128, 0, stream>>>(cb, QCBN, 1024, 64);
        gemm(QXF, 64, cb, 64, QDOT, 1024, 4096, 1024, 64, 1, 1.f, nullptr, 0);
        argmin_update_kernel<<<4096, 128, 0, stream>>>(QDOT, QCBN, cb, QXF, QACC,
                                                       LCB, LCM, 1024, 64, inv_mc);
    }
    nhwc2nchw_kernel<<<(262144 + 255) / 256, 256, 0, stream>>>(QACC, T3, 64, 2048, 262144L);

    conv(T3, 60, T1, 64, 32, 64, 64, 32, 64, 1, 1, 1, 0, 0, 1, 0);        // post_quant
    conv(T1, 62, T0, 64, 32, 64, 256, 32, 64, 3, 3, 1, 1, 1, 1, 0);       // dec_in

    mid(64, 256, 32, 64);                                                 // mid_dec

    // up1: conv_t k4, pad (2,2), lhs_dilation 2  -> (2,256,64,128)
    conv(T0, 90, T3, 256, 32, 64, 256, 64, 128, 4, 4, 1, 2, 2, 2, 0);
    resnet(92, 256, 128, 64, 128, T3, T0);                                // up1.res

    // up2: conv_t k(3,2), pad ((2,3),(1,1)), lhs_dilation 2 -> (2,128,130,256)
    conv(T0, 102, T3, 128, 64, 128, 128, 130, 256, 3, 2, 1, 2, 1, 2, 0);
    {
        long tot = 2L * 128 * 129 * 256;
        crop_rows_kernel<<<(unsigned)((tot + 255) / 256), 256, 0, stream>>>(T3, T0, 128, 130, 129, 256, tot);
    }
    resnet(104, 128, 64, 129, 256, T0, T3);                               // up2.res

    gn(T3, T1, 114, 64, 33024, 1, 0);                                     // dec_norm+silu
    conv(T1, 116, OUT, 64, 129, 256, 2, 129, 256, 3, 3, 1, 1, 1, 1, 0);   // conv_out -> d_out
}
